// SimpleAttention_37160057045086
// MI455X (gfx1250) — matrix operand order repaired and checked by reading, not compiled, not run on hardware
//
#include <hip/hip_runtime.h>

// ---------------------------------------------------------------------------
// SimpleAttention for MI455X (gfx1250, wave32, WMMA + async-to-LDS).
//   x  = img / sqrt(4096)           (fold 1/64 into each bf16 operand)
//   S  = x x^T, diag = -inf         (fp32, staged in d_out scratch)
//   P  = softmax_rows(S)            (bf16)
//   out= P x^T                      (fp32)
// Both GEMMs are NT (C = A * B^T, row-major A,B) -> fragments are plain row
// loads matching CDNA5 WMMA bf16 operand layouts.
//
// GEMM: 256x128 block, 8 wave32 waves in 4(M)x2(N) grid; each wave owns a
// 64x64 region = 4x4 WMMA tiles (16 v_wmma per K-step of 32).
// Staging uses GLOBAL_LOAD_ASYNC_TO_LDS_B128 into double-buffered LDS:
// next slab's loads are issued right after the barrier and complete during
// the current slab's 16 WMMAs (tracked with ASYNCcnt).
// ---------------------------------------------------------------------------

typedef __bf16 bf16_t;
typedef __attribute__((ext_vector_type(16))) __bf16 v16bf;
typedef __attribute__((ext_vector_type(8)))  float  v8f;

#define NDIM   4096
#define BM     256
#define BN     128
#define BK     32
#define LDST   40        // padded LDS row stride (bf16): 80B = 20-bank step
#define LOG2E  1.44269504088896340736f

union FragAB { v16bf v; uint4 u[2]; };
union Pack4  { bf16_t h[4]; uint2 u; };

__device__ __forceinline__ void wait_async0() {
#if __has_builtin(__builtin_amdgcn_s_wait_asynccnt)
    __builtin_amdgcn_s_wait_asynccnt(0);
#else
    asm volatile("s_wait_asynccnt 0x0" ::: "memory");
#endif
}

// Issue the 6 async b128 loads for one K-slab (A: 64B/thread, B: 32B/thread).
// INST_OFFSET applies to both LDS and global addresses (ISA 15.18.3), and our
// slab chunks are offset-congruent, so literal offsets cover all chunks.
__device__ __forceinline__ void stage_async(const bf16_t* gA, unsigned ldsA,
                                            const bf16_t* gB, unsigned ldsB) {
    asm volatile(
        "global_load_async_to_lds_b128 %0, %2, off\n\t"
        "global_load_async_to_lds_b128 %0, %2, off offset:16\n\t"
        "global_load_async_to_lds_b128 %0, %2, off offset:32\n\t"
        "global_load_async_to_lds_b128 %0, %2, off offset:48\n\t"
        "global_load_async_to_lds_b128 %1, %3, off\n\t"
        "global_load_async_to_lds_b128 %1, %3, off offset:16"
        :: "v"(ldsA), "v"(ldsB), "v"(gA), "v"(gB)
        : "memory");
}

// ---------------------------------------------------------------------------
// Kernel 0: xb[i] = bf16(x[i] * (1/64))      (1/64 = 1/sqrt(4096))
// ---------------------------------------------------------------------------
__global__ void __launch_bounds__(256) scale_to_bf16(const float* __restrict__ x,
                                                     bf16_t* __restrict__ xb) {
    const size_t i = ((size_t)blockIdx.x * blockDim.x + threadIdx.x) * 4;
    const float4 v = *(const float4*)(x + i);
    Pack4 p;
    p.h[0] = (bf16_t)(v.x * 0.015625f);
    p.h[1] = (bf16_t)(v.y * 0.015625f);
    p.h[2] = (bf16_t)(v.z * 0.015625f);
    p.h[3] = (bf16_t)(v.w * 0.015625f);
    *(uint2*)(xb + i) = p.u;
}

// ---------------------------------------------------------------------------
// NT GEMM: C[m][n] = sum_k A[m][k] * B[n][k]; bf16 in, fp32 out.
// If diag_mask: C[m][m] = -inf.
// ---------------------------------------------------------------------------
__global__ void __launch_bounds__(256) gemm_nt_bf16(const bf16_t* __restrict__ A,
                                                    const bf16_t* __restrict__ B,
                                                    float* __restrict__ C,
                                                    int diag_mask) {
    __shared__ bf16_t As[2][BM * LDST];   // 2 x 20 KB
    __shared__ bf16_t Bs[2][BN * LDST];   // 2 x 10 KB

    const int tid  = threadIdx.x;
    const int lane = tid & 31;         // wave32
    const int wave = tid >> 5;         // 0..7
    const int wm   = wave >> 1;        // 0..3  -> rows wm*64..+63
    const int wn   = wave & 1;         // 0..1  -> cols wn*64..+63
    const int lrow = lane & 15;
    const int sel  = lane >> 4;        // lane half selects K sub-pattern

    const int blockM = blockIdx.y * BM;
    const int blockN = blockIdx.x * BN;

    const int brow  = tid >> 1;        // 0..127
    const int bhalf = tid & 1;

    // per-thread staging addresses (LDS side, byte addresses, both buffers)
    const unsigned ldsA0 = (unsigned)(size_t)&As[0][tid * LDST];
    const unsigned ldsA1 = (unsigned)(size_t)&As[1][tid * LDST];
    const unsigned ldsB0 = (unsigned)(size_t)&Bs[0][brow * LDST + bhalf * 16];
    const unsigned ldsB1 = (unsigned)(size_t)&Bs[1][brow * LDST + bhalf * 16];

    const bf16_t* gA = A + (size_t)(blockM + tid) * NDIM;
    const bf16_t* gB = B + (size_t)(blockN + brow) * NDIM + bhalf * 16;

    const v8f vzero = {0.f, 0.f, 0.f, 0.f, 0.f, 0.f, 0.f, 0.f};
    v8f acc[4][4];
#pragma unroll
    for (int mt = 0; mt < 4; ++mt)
#pragma unroll
        for (int nt = 0; nt < 4; ++nt) acc[mt][nt] = vzero;

    // prologue: stage slab k=0 into buffer 0
    stage_async(gA, ldsA0, gB, ldsB0);

    for (int k0 = 0; k0 < NDIM; k0 += BK) {
        const int buf = (k0 >> 5) & 1;

        wait_async0();        // own async writes of current buffer landed
        __syncthreads();      // everyone's landed; prev reads of other buf done

        if (k0 + BK < NDIM) { // prefetch next slab into the other buffer
            stage_async(gA + k0 + BK, buf ? ldsA0 : ldsA1,
                        gB + k0 + BK, buf ? ldsB0 : ldsB1);
        }

        // ---- A fragments: lane = M row; K chunks {sel*8..+7},{16+sel*8..+7}
        const bf16_t* __restrict__ Ab = As[buf];
        const bf16_t* __restrict__ Bb = Bs[buf];
        FragAB af[4];
#pragma unroll
        for (int mt = 0; mt < 4; ++mt) {
            const int r = (wm * 64 + mt * 16 + lrow) * LDST;
            af[mt].u[0] = *(const uint4*)&Ab[r + sel * 8];
            af[mt].u[1] = *(const uint4*)&Ab[r + 16 + sel * 8];
        }
        // ---- B fragments ping-pong (breaks WAR chains), 4 wmma each ----
        FragAB bfr[2];
        {
            const int r = (wn * 64 + lrow) * LDST;
            bfr[0].u[0] = *(const uint4*)&Bb[r + sel * 8];
            bfr[0].u[1] = *(const uint4*)&Bb[r + 16 + sel * 8];
        }
#pragma unroll
        for (int nt = 0; nt < 4; ++nt) {
            if (nt < 3) {
                const int r = (wn * 64 + (nt + 1) * 16 + lrow) * LDST;
                bfr[(nt + 1) & 1].u[0] = *(const uint4*)&Bb[r + sel * 8];
                bfr[(nt + 1) & 1].u[1] = *(const uint4*)&Bb[r + 16 + sel * 8];
            }
#pragma unroll
            for (int mt = 0; mt < 4; ++mt)
                acc[mt][nt] = __builtin_amdgcn_wmma_f32_16x16x32_bf16(
                    false, af[mt].v, false, bfr[nt & 1].v,
                    (short)0, acc[mt][nt], false, false);
        }
    }

    // ---- epilogue: C layout = VGPR r -> M = r + sel*8, lane -> N ----
#pragma unroll
    for (int mt = 0; mt < 4; ++mt) {
#pragma unroll
        for (int nt = 0; nt < 4; ++nt) {
            const int m0 = blockM + wm * 64 + mt * 16 + sel * 8;
            const int n  = blockN + wn * 64 + nt * 16 + lrow;
#pragma unroll
            for (int r = 0; r < 8; ++r) {
                float v = acc[mt][nt][r];
                const int m = m0 + r;
                if (diag_mask && (m == n)) v = -__builtin_inff();
                C[(size_t)m * NDIM + n] = v;
            }
        }
    }
}

// ---------------------------------------------------------------------------
// Rowwise softmax: one 256-thread block per row; 16 elems/thread (b128 I/O);
// LDS tree reductions for max and sum; writes bf16 probabilities.
// ---------------------------------------------------------------------------
__global__ void __launch_bounds__(256) softmax_row(const float* __restrict__ S,
                                                   bf16_t* __restrict__ P) {
    __shared__ float red[256];
    const int row = blockIdx.x;
    const int tid = threadIdx.x;
    const float4* s4 = (const float4*)(S + (size_t)row * NDIM);

    float4 v[4];
    float mx = -__builtin_inff();
#pragma unroll
    for (int i = 0; i < 4; ++i) {
        v[i] = s4[tid + i * 256];
        mx = fmaxf(mx, fmaxf(fmaxf(v[i].x, v[i].y), fmaxf(v[i].z, v[i].w)));
    }
    red[tid] = mx;
    __syncthreads();
    for (int off = 128; off > 0; off >>= 1) {
        if (tid < off) red[tid] = fmaxf(red[tid], red[tid + off]);
        __syncthreads();
    }
    mx = red[0];
    __syncthreads();

    float sum = 0.f;
#pragma unroll
    for (int i = 0; i < 4; ++i) {
        v[i].x = exp2f((v[i].x - mx) * LOG2E);   // diag: exp(-inf) = 0
        v[i].y = exp2f((v[i].y - mx) * LOG2E);
        v[i].z = exp2f((v[i].z - mx) * LOG2E);
        v[i].w = exp2f((v[i].w - mx) * LOG2E);
        sum += (v[i].x + v[i].y) + (v[i].z + v[i].w);
    }
    red[tid] = sum;
    __syncthreads();
    for (int off = 128; off > 0; off >>= 1) {
        if (tid < off) red[tid] += red[tid + off];
        __syncthreads();
    }
    const float inv = 1.0f / red[0];

    bf16_t* p = P + (size_t)row * NDIM;
#pragma unroll
    for (int i = 0; i < 4; ++i) {
        Pack4 pk;
        pk.h[0] = (bf16_t)(v[i].x * inv);
        pk.h[1] = (bf16_t)(v[i].y * inv);
        pk.h[2] = (bf16_t)(v[i].z * inv);
        pk.h[3] = (bf16_t)(v[i].w * inv);
        *(uint2*)(p + (tid + i * 256) * 4) = pk.u;
    }
}

// ---------------------------------------------------------------------------
extern "C" void kernel_launch(void* const* d_in, const int* in_sizes, int n_in,
                              void* d_out, int out_size, void* d_ws, size_t ws_size,
                              hipStream_t stream) {
    (void)in_sizes; (void)n_in; (void)out_size; (void)ws_size;

    const float* x   = (const float*)d_in[0];
    float*       out = (float*)d_out;                   // also scratch for S
    bf16_t*      xb   = (bf16_t*)d_ws;                  // 32 MB
    bf16_t*      attn = (bf16_t*)((char*)d_ws + (size_t)NDIM * NDIM * sizeof(bf16_t));

    // 1) scale + convert to bf16
    scale_to_bf16<<<(NDIM * (size_t)NDIM) / (256 * 4), 256, 0, stream>>>(x, xb);

    // 2) S = xb xb^T (carries 1/D), diagonal -> -inf; S lives in d_out scratch
    dim3 grid(NDIM / BN, NDIM / BM);
    gemm_nt_bf16<<<grid, 256, 0, stream>>>(xb, xb, out, 1);

    // 3) P = softmax_rows(S) as bf16
    softmax_row<<<NDIM, 256, 0, stream>>>(out, attn);

    // 4) out = P xb^T
    gemm_nt_bf16<<<grid, 256, 0, stream>>>(attn, xb, out, 0);
}
